// SimpleNet_58256936403066
// MI455X (gfx1250) — matrix operand order repaired and checked by reading, not compiled, not run on hardware
//
#include <hip/hip_runtime.h>

#define N_VV    100000
#define N_CV    100000
#define N_EDGES 800000
#define NT_TILES (N_EDGES / 16)        // 50000 16-edge tiles
#define EDGE_BLOCKS 625                // 625 blocks * 8 waves * 10 tiles = 50000
#define H       64
#define BN_EPS  1e-5f

typedef __attribute__((ext_vector_type(16))) __bf16 v16bf;
typedef __attribute__((ext_vector_type(8)))  float  v8f;

union V16 { v16bf v; uint4 q[2]; unsigned uw[8]; };

__device__ __forceinline__ unsigned short f2bf(float f) {
  unsigned u = __float_as_uint(f);
  unsigned r = u + 0x7FFFu + ((u >> 16) & 1u);   // round-to-nearest-even
  return (unsigned short)(r >> 16);
}

// ---------------------------------------------------------------- utilities
__global__ void zero_f32(float* __restrict__ p, int n) {
  int i = blockIdx.x * blockDim.x + threadIdx.x;
  if (i < n) p[i] = 0.0f;
}

// Pack W1 [128x64] and W2 [64x64] (row-major, B = K x N) into WMMA B-fragment
// order: frag f = kt*4+nt ; entry [f][lane][v] = pack(bf16(W[k0][n]), bf16(W[k0+1][n]))
// with n = nt*16 + (lane&15), k0 = kt*32 + (lane>=16 ? 16 : 0) + 2v.
__global__ void prep_weights(const float* __restrict__ w1, const float* __restrict__ w2,
                             unsigned* __restrict__ w1p, unsigned* __restrict__ w2p) {
  int p = blockIdx.x * blockDim.x + threadIdx.x;   // 6144 threads total
  if (p < 4096) {                                   // W1: 16 frags * 32 lanes * 8
    int f = p >> 8, rem = p & 255;
    int lane = rem >> 3, v = rem & 7;
    int kt = f >> 2, nt = f & 3;
    int n  = nt * 16 + (lane & 15);
    int k0 = kt * 32 + ((lane & 16) ? 16 : 0) + 2 * v;
    unsigned lo = f2bf(w1[k0 * H + n]);
    unsigned hi = f2bf(w1[(k0 + 1) * H + n]);
    w1p[p] = lo | (hi << 16);
  } else if (p < 6144) {                            // W2: 8 frags * 32 lanes * 8
    int q = p - 4096;
    int f = q >> 8, rem = q & 255;
    int lane = rem >> 3, v = rem & 7;
    int kt = f >> 2, nt = f & 3;
    int n  = nt * 16 + (lane & 15);
    int k0 = kt * 32 + ((lane & 16) ? 16 : 0) + 2 * v;
    unsigned lo = f2bf(w2[k0 * H + n]);
    unsigned hi = f2bf(w2[(k0 + 1) * H + n]);
    w2p[q] = lo | (hi << 16);
  }
}

// ------------------------------------------------- node encoder (2->64->64)
__global__ void encode_nodes(const float* __restrict__ x,
                             const float* __restrict__ w1, const float* __restrict__ b1,
                             const float* __restrict__ w2, const float* __restrict__ b2,
                             unsigned short* __restrict__ outb, int n_nodes) {
  __shared__ float h1s[4][H];
  int g = threadIdx.x >> 6;         // 4 nodes per 256-thread block
  int t = threadIdx.x & 63;
  int node = blockIdx.x * 4 + g;
  float x0 = 0.f, x1 = 0.f;
  if (node < n_nodes) { x0 = x[node * 2]; x1 = x[node * 2 + 1]; }
  float h = x0 * w1[t] + x1 * w1[H + t] + b1[t];
  h1s[g][t] = h > 0.f ? h : 0.f;
  __syncthreads();
  if (node < n_nodes) {
    float acc = b2[t];
#pragma unroll 8
    for (int k = 0; k < H; ++k) acc += h1s[g][k] * w2[k * H + t];
    outb[node * H + t] = f2bf(acc);
  }
}

// --------------------------------------------------------- edge MLP (WMMA)
// One wave = a strided sequence of 16-edge tiles. Weights staged in LDS once
// per block; layer-1 A fragments gathered via global_load_b128 clauses.
// PASS 1: accumulate BN sum/sumsq.  PASS 2: fused BN + atomic scatter-mean.
template <int PASS>
__global__ __launch_bounds__(256) void edge_mlp(
    const int* __restrict__ eidx,                 // [2, E]: row0 = j(vv), row1 = i(cv)
    const unsigned short* __restrict__ tgtb,      // [N_CV, 64] bf16
    const unsigned short* __restrict__ srcb,      // [N_VV, 64] bf16
    const unsigned* __restrict__ w1p, const unsigned* __restrict__ w2p,
    const float* __restrict__ b1, const float* __restrict__ b2,
    float* __restrict__ gsum, float* __restrict__ gssq,
    const float* __restrict__ scale, const float* __restrict__ shift,
    float* __restrict__ outp, float* __restrict__ cnt) {
  __shared__ __align__(16) unsigned w1s[16 * 32 * 8];        // 16 KB
  __shared__ __align__(16) unsigned w2s[8 * 32 * 8];         // 8 KB
  __shared__ __align__(16) unsigned short h1buf[8][16 * H];  // 16 KB, per-wave
  __shared__ int ibuf[8][16];
  __shared__ float bsum[H], bssq[H];

  const int tid  = threadIdx.x;
  const int lane = tid & 31;
  const int w    = tid >> 5;

  // stage weight fragments into LDS (shared by all 8 waves)
  {
    const uint4* s1 = (const uint4*)w1p;  uint4* d1 = (uint4*)w1s;
#pragma unroll
    for (int i = 0; i < 4; ++i) d1[tid + 256 * i] = s1[tid + 256 * i];
    const uint4* s2 = (const uint4*)w2p;  uint4* d2 = (uint4*)w2s;
#pragma unroll
    for (int i = 0; i < 2; ++i) d2[tid + 256 * i] = s2[tid + 256 * i];
  }
  if (PASS == 1 && tid < H) { bsum[tid] = 0.f; bssq[tid] = 0.f; }
  __syncthreads();

  const int  e      = lane & 15;
  const bool hiHalf = lane >= 16;
  const int  Lm     = lane & 15;

  // per-lane feature-column constants (feature n = nt*16 + Lm)
  float b1v[4], b2v[4], scv[4], shv[4];
#pragma unroll
  for (int nt = 0; nt < 4; ++nt) {
    int n = nt * 16 + Lm;
    b1v[nt] = b1[n];
    b2v[nt] = b2[n];
    if (PASS == 2) { scv[nt] = scale[n]; shv[nt] = shift[n]; }
    else           { scv[nt] = 0.f;      shv[nt] = 0.f; }
  }

  const int tile0 = blockIdx.x * 8 + w;
  const int step  = EDGE_BLOCKS * 8;                 // 5000 wave slots

  for (int tile = tile0; tile < NT_TILES; tile += step) {
    const int ge  = tile * 16 + e;
    const int jn  = eidx[ge];                        // vv (src) node
    const int in_ = eidx[N_EDGES + ge];              // cv (tgt) node
    if (PASS == 2 && !hiHalf) ibuf[w][e] = in_;

    const unsigned short* irow = tgtb + (size_t)in_ * H;
    const unsigned short* jrow = srcb + (size_t)jn  * H;

    // ---- preload all layer-1 A fragments (one clause of 8 b128 loads) ----
    V16 A1[4];
#pragma unroll
    for (int kt = 0; kt < 4; ++kt) {
      const unsigned short* row = (kt < 2) ? irow : jrow;   // concat split at K=64
      int koff = (kt & 1) * 32 + (hiHalf ? 16 : 0);
      const uint4* ap = (const uint4*)(row + koff);
      A1[kt].q[0] = ap[0];          // rel K 0..7
      A1[kt].q[1] = ap[1];          // rel K 8..15
    }

    // ---- layer 1: [16x128] @ [128x64], B fragments from LDS ----
    v8f acc[4];
#pragma unroll
    for (int nt = 0; nt < 4; ++nt) acc[nt] = v8f{0.f,0.f,0.f,0.f,0.f,0.f,0.f,0.f};
#pragma unroll
    for (int kt = 0; kt < 4; ++kt) {
#pragma unroll
      for (int nt = 0; nt < 4; ++nt) {
        V16 B;
        const uint4* bp = (const uint4*)(w1s + ((kt * 4 + nt) * 256 + lane * 8));
        B.q[0] = bp[0]; B.q[1] = bp[1];
        acc[nt] = __builtin_amdgcn_wmma_f32_16x16x32_bf16(
            false, A1[kt].v, false, B.v, (short)0, acc[nt], false, false);
      }
    }

    // bias + ReLU + repack h1 (bf16, [edge][feat]) into per-wave LDS
#pragma unroll
    for (int nt = 0; nt < 4; ++nt) {
#pragma unroll
      for (int r = 0; r < 8; ++r) {
        float v = acc[nt][r] + b1v[nt];
        v = v > 0.f ? v : 0.f;
        int e2 = r + (hiHalf ? 8 : 0);               // D layout: M = r (+8 hi lanes)
        h1buf[w][e2 * H + nt * 16 + Lm] = f2bf(v);
      }
    }
    asm volatile("s_wait_dscnt 0" ::: "memory");     // same-wave cross-lane LDS hazard

    // ---- layer 2: [16x64] @ [64x64] ----
    v8f acc2[4];
#pragma unroll
    for (int nt = 0; nt < 4; ++nt) acc2[nt] = v8f{0.f,0.f,0.f,0.f,0.f,0.f,0.f,0.f};
#pragma unroll
    for (int kt = 0; kt < 2; ++kt) {
      V16 A;
      const uint4* hp = (const uint4*)(&h1buf[w][e * H + kt * 32 + (hiHalf ? 16 : 0)]);
      A.q[0] = hp[0];
      A.q[1] = hp[1];
#pragma unroll
      for (int nt = 0; nt < 4; ++nt) {
        V16 B;
        const uint4* bp = (const uint4*)(w2s + ((kt * 4 + nt) * 256 + lane * 8));
        B.q[0] = bp[0]; B.q[1] = bp[1];
        acc2[nt] = __builtin_amdgcn_wmma_f32_16x16x32_bf16(
            false, A.v, false, B.v, (short)0, acc2[nt], false, false);
      }
    }

    if (PASS == 1) {
#pragma unroll
      for (int nt = 0; nt < 4; ++nt) {
        float s = 0.f, ss = 0.f;
#pragma unroll
        for (int r = 0; r < 8; ++r) {
          float m = acc2[nt][r] + b2v[nt];
          m = m > 0.f ? m : 0.f;
          s += m; ss += m * m;
        }
        atomicAdd(&bsum[nt * 16 + Lm], s);
        atomicAdd(&bssq[nt * 16 + Lm], ss);
      }
    } else {
      if (!hiHalf) atomicAdd(&cnt[in_], 1.0f);
      asm volatile("s_wait_dscnt 0" ::: "memory");   // ibuf visibility across lanes
#pragma unroll
      for (int nt = 0; nt < 4; ++nt) {
#pragma unroll
        for (int r = 0; r < 8; ++r) {
          float m = acc2[nt][r] + b2v[nt];
          m = m > 0.f ? m : 0.f;
          float v = m * scv[nt] + shv[nt];
          int e2 = r + (hiHalf ? 8 : 0);
          int node = ibuf[w][e2];
          atomicAdd(&outp[(size_t)node * H + nt * 16 + Lm], v);
        }
      }
    }
  }

  if (PASS == 1) {
    __syncthreads();
    if (tid < H) {
      atomicAdd(&gsum[tid], bsum[tid]);
      atomicAdd(&gssq[tid], bssq[tid]);
    }
  }
}

// ------------------------------------------------ BN fold + final mean-div
__global__ void finalize_stats(const float* __restrict__ gsum, const float* __restrict__ gssq,
                               const float* __restrict__ gamma, const float* __restrict__ beta,
                               float* __restrict__ scale, float* __restrict__ shift) {
  int t = threadIdx.x;
  if (t >= H) return;
  const float inv_e = 1.0f / (float)N_EDGES;
  float mu  = gsum[t] * inv_e;
  float var = gssq[t] * inv_e - mu * mu;
  float sc  = rsqrtf(var + BN_EPS) * gamma[t];
  scale[t] = sc;
  shift[t] = beta[t] - mu * sc;
}

__global__ void finalize_out(float* __restrict__ out, const float* __restrict__ cnt) {
  int p = blockIdx.x * blockDim.x + threadIdx.x;
  if (p < N_CV * H) {
    float c = cnt[p >> 6];
    c = c > 1.f ? c : 1.f;
    out[p] = out[p] / c;
  }
}

// ----------------------------------------------------------------- launcher
extern "C" void kernel_launch(void* const* d_in, const int* in_sizes, int n_in,
                              void* d_out, int out_size, void* d_ws, size_t ws_size,
                              hipStream_t stream) {
  const float* vvx   = (const float*)d_in[0];
  const float* cvx   = (const float*)d_in[1];
  const int*   eidx  = (const int*)d_in[2];
  const float* vv_w1 = (const float*)d_in[3];
  const float* vv_b1 = (const float*)d_in[4];
  const float* vv_w2 = (const float*)d_in[5];
  const float* vv_b2 = (const float*)d_in[6];
  const float* cv_w1 = (const float*)d_in[7];
  const float* cv_b1 = (const float*)d_in[8];
  const float* cv_w2 = (const float*)d_in[9];
  const float* cv_b2 = (const float*)d_in[10];
  const float* nn_w1 = (const float*)d_in[11];
  const float* nn_b1 = (const float*)d_in[12];
  const float* nn_w2 = (const float*)d_in[13];
  const float* nn_b2 = (const float*)d_in[14];
  const float* bn_g  = (const float*)d_in[15];
  const float* bn_b  = (const float*)d_in[16];

  char* ws = (char*)d_ws;
  size_t off = 0;
  auto carve = [&](size_t bytes) -> char* {
    char* p = ws + off;
    off = (off + bytes + 255) & ~(size_t)255;
    return p;
  };
  unsigned short* srcb = (unsigned short*)carve((size_t)N_VV * H * 2);
  unsigned short* tgtb = (unsigned short*)carve((size_t)N_CV * H * 2);
  unsigned*       w1p  = (unsigned*)carve(16 * 32 * 8 * 4);
  unsigned*       w2p  = (unsigned*)carve(8 * 32 * 8 * 4);
  float*          stats = (float*)carve((size_t)(2 * H + N_CV) * 4);  // sum|sumsq|cnt
  float*          gsum = stats;
  float*          gssq = stats + H;
  float*          cnt  = stats + 2 * H;
  float*          scl  = (float*)carve(H * 4);
  float*          shf  = (float*)carve(H * 4);

  float* out = (float*)d_out;

  zero_f32<<<(N_CV * H + 255) / 256, 256, 0, stream>>>(out, N_CV * H);
  zero_f32<<<(2 * H + N_CV + 255) / 256, 256, 0, stream>>>(stats, 2 * H + N_CV);
  prep_weights<<<24, 256, 0, stream>>>(nn_w1, nn_w2, w1p, w2p);
  encode_nodes<<<(N_VV + 3) / 4, 256, 0, stream>>>(vvx, vv_w1, vv_b1, vv_w2, vv_b2, srcb, N_VV);
  encode_nodes<<<(N_CV + 3) / 4, 256, 0, stream>>>(cvx, cv_w1, cv_b1, cv_w2, cv_b2, tgtb, N_CV);
  edge_mlp<1><<<EDGE_BLOCKS, 256, 0, stream>>>(eidx, tgtb, srcb, w1p, w2p, nn_b1, nn_b2,
                                               gsum, gssq, nullptr, nullptr, nullptr, nullptr);
  finalize_stats<<<1, 64, 0, stream>>>(gsum, gssq, bn_g, bn_b, scl, shf);
  edge_mlp<2><<<EDGE_BLOCKS, 256, 0, stream>>>(eidx, tgtb, srcb, w1p, w2p, nn_b1, nn_b2,
                                               nullptr, nullptr, scl, shf, out, cnt);
  finalize_out<<<(N_CV * H + 255) / 256, 256, 0, stream>>>(out, cnt);
}
